// RealNVP_9191230013833
// MI455X (gfx1250) — matrix operand order repaired and checked by reading, not compiled, not run on hardware
//
#include <hip/hip_runtime.h>
#include <hip/hip_bf16.h>

// ---------------- problem constants ----------------
#define NLAYERS 8
#define DLAT    4
#define WFLEN   9198          // waveform length (K of big GEMM)
#define HDIM    16
#define BATCH   16384
#define NEG_SLOPE 0.01f

#define KSTEP   32            // WMMA K per step (f16)
#define KSTEPS  288           // ceil(9198/32) -> padded K = 9216
#define NCH     256           // 8 layers * 2 nets * 16 hidden = channels (M of GEMM)

typedef __attribute__((ext_vector_type(16))) _Float16 v16h;
typedef __attribute__((ext_vector_type(8)))  float    v8f;

// Workspace layout:
//   Acat (fragment-major packed f16 W1 waveform-part): 16 * 288 * 32 * 16 halves
//   Ct   (f32, [NCH][BATCH], i.e. transposed GEMM result)
#define ACAT_ELEMS ((size_t)16 * KSTEPS * 32 * 16)     // 2,359,296 halves
#define ACAT_BYTES (ACAT_ELEMS * 2)                    // 4,718,592 B (256-aligned)
#define CT_ELEMS   ((size_t)NCH * BATCH)               // 4,194,304 floats

// =====================================================================
// Kernel 1: pack W1 waveform-rows (both nets, all layers) into a
// per-lane A-fragment layout used with V_WMMA_F32_16X16X32_F16 (contiguous k, same as B), converted to f16.
// Fragment index layout: (((mt*KSTEPS)+ks)*32 + lane)*16 + j
//   lanes 0-15 : j0-15 -> K k0+0..15
//   lanes 16-31: j0-15 -> K k0+16..31
// Channel c = mt*16 + (lane&15);  c = layer*32 + net*16 + h
// =====================================================================
__global__ void prep_pack_A(const float* __restrict__ sW1,
                            const float* __restrict__ tW1,
                            _Float16* __restrict__ Acat) {
  size_t idx = (size_t)blockIdx.x * 256 + threadIdx.x;
  if (idx >= ACAT_ELEMS) return;
  int j    = (int)(idx & 15);
  int lane = (int)((idx >> 4) & 31);
  int ks   = (int)((idx >> 9) % KSTEPS);
  int mt   = (int)(idx / ((size_t)KSTEPS * 512));
  int c    = mt * 16 + (lane & 15);
  int li   = c >> 5;
  int net  = (c >> 4) & 1;
  int h    = c & 15;
  int kl   = j + ((lane & 16) ? 16 : 0);
  int K    = ks * KSTEP + kl;
  float v  = 0.0f;
  if (K < WFLEN) {
    const float* W1 = net ? tW1 : sW1;
    v = W1[((size_t)li * (DLAT + WFLEN) + DLAT + K) * HDIM + h];
  }
  Acat[idx] = (_Float16)v;
}

// =====================================================================
// Kernel 2: big GEMM  Ct[256][16384] = Wcat^T (f16) x waveform^T (f16)
// Grid: 64 WGs (256 batch cols each). Block: 512 threads = 16 waves.
// Wave (wm,wn) computes a 64x64 sub-tile = 4x4 WMMA tiles, K streamed
// through double-buffered LDS with in-flight f32->f16 conversion.
// =====================================================================
__device__ __forceinline__ v16h load_wf_frag(const float* __restrict__ wf,
                                             int n0, int tid, int ks) {
  int n = tid >> 1, half = tid & 1;
  int k0 = ks * KSTEP + half * 16;
  const float* row = wf + (size_t)(n0 + n) * WFLEN + k0;
  v16h r;
  if (k0 + 16 <= WFLEN) {
    const float2* p = (const float2*)row;   // row offset is 8B aligned
#pragma unroll
    for (int q = 0; q < 8; ++q) {
      float2 v = p[q];
      r[2 * q]     = (_Float16)v.x;
      r[2 * q + 1] = (_Float16)v.y;
    }
  } else {
#pragma unroll
    for (int q = 0; q < 16; ++q) {
      float v = (k0 + q < WFLEN) ? row[q] : 0.0f;
      r[q] = (_Float16)v;
    }
  }
  return r;
}

__device__ __forceinline__ void store_wf_frag(_Float16* lds, int tid, v16h r) {
  int n = tid >> 1, half = tid & 1;
  int nt = n >> 4;                          // N-tile within block (0..15)
  int l  = (n & 15) | (half << 4);          // B-matrix lane for this data
  *(v16h*)(lds + (((nt * 32) + l) << 4)) = r;
}

__global__ void __launch_bounds__(512)
gemm_wmma(const float* __restrict__ wf,
          const _Float16* __restrict__ Acat,
          float* __restrict__ Ct) {
  __shared__ __align__(32) _Float16 ldsB[2][16 * 32 * 16];   // 2 x 16KB

  const int tid  = threadIdx.x;
  const int lane = tid & 31;
  const int wave = tid >> 5;
  const int wm   = wave & 3;      // M quadrant (4 tiles each)
  const int wn   = wave >> 2;     // N quadrant (4 tiles each)
  const int n0   = blockIdx.x * 256;

  v8f acc[4][4];
#pragma unroll
  for (int a = 0; a < 4; ++a)
#pragma unroll
    for (int nb = 0; nb < 4; ++nb)
      acc[a][nb] = (v8f){0.f, 0.f, 0.f, 0.f, 0.f, 0.f, 0.f, 0.f};

  const v16h* Av = (const v16h*)Acat;

  // prologue: stage K-step 0
  v16h breg = load_wf_frag(wf, n0, tid, 0);
  store_wf_frag(&ldsB[0][0], tid, breg);
  __syncthreads();

  int buf = 0;
  for (int ks = 0; ks < KSTEPS; ++ks) {
    v16h nxt;
    const bool have = (ks + 1 < KSTEPS);
    if (have) nxt = load_wf_frag(wf, n0, tid, ks + 1);   // overlap with WMMA

    v16h afr[4], bfr[4];
#pragma unroll
    for (int a = 0; a < 4; ++a)
      afr[a] = Av[(size_t)((wm * 4 + a) * KSTEPS + ks) * 32 + lane];
#pragma unroll
    for (int nb = 0; nb < 4; ++nb)
      bfr[nb] = *(const v16h*)&ldsB[buf][(((wn * 4 + nb) * 32) + lane) << 4];

#pragma unroll
    for (int a = 0; a < 4; ++a)
#pragma unroll
      for (int nb = 0; nb < 4; ++nb)
        acc[a][nb] = __builtin_amdgcn_wmma_f32_16x16x32_f16(
            false, afr[a], false, bfr[nb], (short)0, acc[a][nb], false, false);

    if (have) store_wf_frag(&ldsB[buf ^ 1][0], tid, nxt);
    __syncthreads();
    buf ^= 1;
  }

  // D layout: VGPR r -> (M=r, N=lane) lanes 0-15 ; (M=r+8, N=lane-16) lanes 16-31
#pragma unroll
  for (int a = 0; a < 4; ++a) {
    int cb = (wm * 4 + a) * 16 + ((lane & 16) ? 8 : 0);
#pragma unroll
    for (int nb = 0; nb < 4; ++nb) {
      int col = n0 + (wn * 4 + nb) * 16 + (lane & 15);
#pragma unroll
      for (int r = 0; r < 8; ++r)
        Ct[(size_t)(cb + r) * BATCH + col] = acc[a][nb][r];
    }
  }
}

// =====================================================================
// Kernel 3: sequential RealNVP inverse epilogue. One thread per batch row.
// All small weights staged in LDS. Ct reads are coalesced (thread = batch).
// =====================================================================
__device__ __forceinline__ float leaky(float v) {
  return v >= 0.0f ? v : NEG_SLOPE * v;
}

__global__ void __launch_bounds__(256)
nvp_epilogue(const float* __restrict__ x,    const float* __restrict__ mask,
             const float* __restrict__ sW1,  const float* __restrict__ sW2,
             const float* __restrict__ sW3,  const float* __restrict__ tW1,
             const float* __restrict__ tW2,  const float* __restrict__ tW3,
             const float* __restrict__ sb1,  const float* __restrict__ sb2,
             const float* __restrict__ sb3,  const float* __restrict__ tb1,
             const float* __restrict__ tb2,  const float* __restrict__ tb3,
             const float* __restrict__ Ct,   float* __restrict__ out) {
  __shared__ float lw1[2][NLAYERS][DLAT][HDIM];     // z-part of W1
  __shared__ float lw2[2][NLAYERS][HDIM][HDIM];
  __shared__ float lw3[2][NLAYERS][HDIM][DLAT];
  __shared__ float lb1[2][NLAYERS][HDIM];
  __shared__ float lb2[2][NLAYERS][HDIM];
  __shared__ float lb3[2][NLAYERS][DLAT];
  __shared__ float lmask[NLAYERS][DLAT];

  const int tid = threadIdx.x;

  // z-part of W1 is not contiguous in the source (rows 0..3 of 9202)
  for (int idx = tid; idx < NLAYERS * DLAT * HDIM; idx += 256) {
    int li = idx / (DLAT * HDIM);
    int d  = (idx / HDIM) % DLAT;
    int h  = idx % HDIM;
    size_t src = ((size_t)li * (DLAT + WFLEN) + d) * HDIM + h;
    lw1[0][li][d][h] = sW1[src];
    lw1[1][li][d][h] = tW1[src];
  }
  for (int idx = tid; idx < NLAYERS * HDIM * HDIM; idx += 256) {
    (&lw2[0][0][0][0])[idx] = sW2[idx];
    (&lw2[1][0][0][0])[idx] = tW2[idx];
  }
  for (int idx = tid; idx < NLAYERS * HDIM * DLAT; idx += 256) {
    (&lw3[0][0][0][0])[idx] = sW3[idx];
    (&lw3[1][0][0][0])[idx] = tW3[idx];
  }
  for (int idx = tid; idx < NLAYERS * HDIM; idx += 256) {
    (&lb1[0][0][0])[idx] = sb1[idx];
    (&lb1[1][0][0])[idx] = tb1[idx];
    (&lb2[0][0][0])[idx] = sb2[idx];
    (&lb2[1][0][0])[idx] = tb2[idx];
  }
  for (int idx = tid; idx < NLAYERS * DLAT; idx += 256) {
    (&lb3[0][0][0])[idx] = sb3[idx];
    (&lb3[1][0][0])[idx] = tb3[idx];
    (&lmask[0][0])[idx]  = mask[idx];
  }
  __syncthreads();

  const size_t b = (size_t)blockIdx.x * 256 + tid;
  float z[DLAT];
#pragma unroll
  for (int d = 0; d < DLAT; ++d) z[d] = x[b * DLAT + d];
  float ld = 0.0f;

  for (int li = NLAYERS - 1; li >= 0; --li) {
    float m[DLAT], zm[DLAT];
#pragma unroll
    for (int d = 0; d < DLAT; ++d) { m[d] = lmask[li][d]; zm[d] = m[d] * z[d]; }

    float o[2][DLAT];
#pragma unroll
    for (int net = 0; net < 2; ++net) {
      const float* ctbase = Ct + (size_t)(li * 32 + net * 16) * BATCH + b;
      float h1[HDIM];
#pragma unroll
      for (int h = 0; h < HDIM; ++h) {
        float a = ctbase[(size_t)h * BATCH] + lb1[net][li][h];
#pragma unroll
        for (int d = 0; d < DLAT; ++d) a += zm[d] * lw1[net][li][d][h];
        h1[h] = leaky(a);
      }
      float h2[HDIM];
#pragma unroll
      for (int j = 0; j < HDIM; ++j) {
        float a = lb2[net][li][j];
#pragma unroll
        for (int h = 0; h < HDIM; ++h) a += h1[h] * lw2[net][li][h][j];
        h2[j] = leaky(a);
      }
#pragma unroll
      for (int d = 0; d < DLAT; ++d) {
        float a = lb3[net][li][d];
#pragma unroll
        for (int h = 0; h < HDIM; ++h) a += h2[h] * lw3[net][li][h][d];
        o[net][d] = a;
      }
    }
#pragma unroll
    for (int d = 0; d < DLAT; ++d) {
      float im = 1.0f - m[d];
      float s  = tanhf(o[0][d]) * im;
      float t  = o[1][d] * im;
      z[d] = im * (z[d] - t) * expf(-s) + zm[d];
      ld -= s;
    }
  }

#pragma unroll
  for (int d = 0; d < DLAT; ++d) out[b * DLAT + d] = z[d];
  out[(size_t)BATCH * DLAT + b] = ld;
}

// =====================================================================
extern "C" void kernel_launch(void* const* d_in, const int* in_sizes, int n_in,
                              void* d_out, int out_size, void* d_ws, size_t ws_size,
                              hipStream_t stream) {
  const float* x    = (const float*)d_in[0];
  const float* wf   = (const float*)d_in[1];
  const float* mask = (const float*)d_in[2];
  const float* sW1  = (const float*)d_in[3];
  const float* sW2  = (const float*)d_in[4];
  const float* sW3  = (const float*)d_in[5];
  const float* tW1  = (const float*)d_in[6];
  const float* tW2  = (const float*)d_in[7];
  const float* tW3  = (const float*)d_in[8];
  const float* sb1  = (const float*)d_in[9];
  const float* sb2  = (const float*)d_in[10];
  const float* sb3  = (const float*)d_in[11];
  const float* tb1  = (const float*)d_in[12];
  const float* tb2  = (const float*)d_in[13];
  const float* tb3  = (const float*)d_in[14];
  float* out = (float*)d_out;

  _Float16* Acat = (_Float16*)d_ws;
  float*    Ct   = (float*)((char*)d_ws + ACAT_BYTES);

  // 1) pack/convert W1 waveform-part into WMMA A-fragment layout (f16)
  prep_pack_A<<<(unsigned)((ACAT_ELEMS + 255) / 256), 256, 0, stream>>>(sW1, tW1, Acat);

  // 2) big GEMM: Ct[256][16384] = Wcat^T x waveform^T  (f16 WMMA, f32 acc)
  gemm_wmma<<<BATCH / 256, 512, 0, stream>>>(wf, Acat, Ct);

  // 3) sequential coupling-layer epilogue (z + log_det)
  nvp_epilogue<<<BATCH / 256, 256, 0, stream>>>(
      x, mask, sW1, sW2, sW3, tW1, tW2, tW3,
      sb1, sb2, sb3, tb1, tb2, tb3, Ct, out);
}
